// BenesBlock_54211077210678
// MI455X (gfx1250) — compile-verified, matrix operand order checked by reading, not run on hardware
//
#include <hip/hip_runtime.h>
#include <hip/hip_bf16.h>
#include <math.h>

// ---------------------------------------------------------------------------
// BenesBlock forward for MI455X (gfx1250, wave32, WMMA bf16)
//
// 22 switch layers; per layer:
//   gemm1 (16384x384x768, bf16 WMMA, f32 accum) -> seq-axis LN stats ->
//   normalize+tanh-GELU+cvt -> gemm2 (16384x768x384) fused with residual mix
//   and the quaternary-rotation permutation scatter.
// Weight tiles are staged in LDS per workgroup via the gfx1250 async-to-LDS
// engine (ASYNCcnt); A fragments are register double-buffered.
// ---------------------------------------------------------------------------

typedef __attribute__((ext_vector_type(16))) __bf16 v16bf;
typedef __attribute__((ext_vector_type(8)))  float  v8f;
typedef int v4i __attribute__((__vector_size__(4 * sizeof(int))));

#define CAND_W_F 0.10897247358851683f   // sqrt(1-0.9^2)*0.25

#if defined(__HIP_DEVICE_COMPILE__) && __has_builtin(__builtin_amdgcn_global_load_async_to_lds_b128)
#define ASYNC_LDS_COPY 1
#endif

__device__ __forceinline__ unsigned short f2bf(float f) {
    union { float f; unsigned u; } v; v.f = f;
    unsigned u = v.u;
    u += 0x7FFFu + ((u >> 16) & 1u);     // round-to-nearest-even
    return (unsigned short)(u >> 16);
}

// z-order: flat[p] = row*64+col with row=odd bit-pairs, col=even bit-pairs
__device__ __forceinline__ int zflat(int p) {
    int row = 0, col = 0;
#pragma unroll
    for (int b = 0; b < 6; ++b) {
        int q = (p >> (2 * b)) & 3;
        row |= ((q >> 1) & 1) << b;
        col |= (q & 1) << b;
    }
    return row * 64 + col;
}

// cooperative copy of a contiguous [64 x K] bf16 weight strip into LDS
// (rows of the transposed weight are contiguous, so this is a flat copy)
__device__ __forceinline__ void stage_weights(const unsigned short* __restrict__ src,
                                              unsigned short* sB, int K, int tid) {
    const int CH = 64 * K / 8;  // number of 16-byte chunks (exact multiple of 256)
#if ASYNC_LDS_COPY
    typedef __attribute__((address_space(1))) v4i g_v4i;
    typedef __attribute__((address_space(3))) v4i l_v4i;
    g_v4i* gsrc = (g_v4i*)src;
    l_v4i* ldst = (l_v4i*)sB;
    for (int i = tid; i < CH; i += 256)
        __builtin_amdgcn_global_load_async_to_lds_b128(gsrc + i, ldst + i, 0, 0);
#if __has_builtin(__builtin_amdgcn_s_wait_asynccnt)
    __builtin_amdgcn_s_wait_asynccnt(0);
#else
    asm volatile("s_wait_asynccnt 0" ::: "memory");
#endif
#else
    for (int i = tid; i < CH; i += 256)
        ((uint4*)sB)[i] = ((const uint4*)src)[i];
#endif
    __syncthreads();
}

// ---------------- weight prep: f32 row-major -> bf16 transposed ------------
__global__ __launch_bounds__(256) void prep_w1(const float* __restrict__ W1,
                                               unsigned short* __restrict__ W1T) {
    int t = blockIdx.x * blockDim.x + threadIdx.x;
    const int TOT = 6 * 384 * 768;
    if (t >= TOT) return;
    int n = t % 768;
    int k = (t / 768) % 384;
    int g = t / (768 * 384);
    W1T[((size_t)g * 768 + n) * 384 + k] = f2bf(W1[t]);
}
__global__ __launch_bounds__(256) void prep_w2(const float* __restrict__ W2,
                                               unsigned short* __restrict__ W2T) {
    int t = blockIdx.x * blockDim.x + threadIdx.x;
    const int TOT = 6 * 768 * 384;
    if (t >= TOT) return;
    int n = t % 384;
    int k = (t / 384) % 768;
    int g = t / (384 * 768);
    W2T[((size_t)g * 384 + n) * 768 + k] = f2bf(W2[t]);
}

// ---------------- initial Z-order gather: x -> seq (f32 + bf16) ------------
__global__ __launch_bounds__(256) void init_perm(const float* __restrict__ x,
                                                 float* __restrict__ seqF,
                                                 unsigned short* __restrict__ seqB) {
    int t = blockIdx.x * blockDim.x + threadIdx.x;
    const int TOT = 16 * 4096 * 96;
    if (t >= TOT) return;
    int u = t % 96;
    int p = (t / 96) % 4096;
    int b = t / (96 * 4096);
    float val = x[((size_t)b * 4096 + zflat(p)) * 96 + u];
    size_t di = (size_t)b * (1024 * 384) + (size_t)(p >> 2) * 384 + (p & 3) * 96 + u;
    seqF[di] = val;
    seqB[di] = f2bf(val);
}

// ---------------- GEMM1: F[16384,768] = seq_bf16[16384,384] @ W1 -----------
// block = 8 waves: weight strip (64 cols x 384 K) staged in LDS, each wave
// computes a 16x64 strip; A fragment register double-buffered.
__global__ __launch_bounds__(256) void gemm1_bf16(const unsigned short* __restrict__ A,
                                                  const unsigned short* __restrict__ Bt,
                                                  float* __restrict__ Fo) {
    extern __shared__ unsigned short sB[];               // [64][384]
    const int K = 384, Nn = 768, NT = Nn / 64;           // 12 n-groups
    int tid  = threadIdx.x;
    int lane = tid & 31, wv = tid >> 5;
    int mt8  = blockIdx.x / NT, ng = blockIdx.x % NT;
    int col0 = ng * 64;

    stage_weights(Bt + (size_t)col0 * K, sB, K, tid);

    int mt   = mt8 * 8 + wv;
    int row0 = mt * 16;
    int laneM = lane & 15, laneH = lane >> 4;

    const unsigned short* ap  = A + (size_t)(row0 + laneM) * K + laneH * 16;
    const unsigned short* sb0 = sB + (size_t)( 0 + laneM) * K + laneH * 16;
    const unsigned short* sb1 = sB + (size_t)(16 + laneM) * K + laneH * 16;
    const unsigned short* sb2 = sB + (size_t)(32 + laneM) * K + laneH * 16;
    const unsigned short* sb3 = sB + (size_t)(48 + laneM) * K + laneH * 16;

    v8f c0 = {}, c1 = {}, c2 = {}, c3 = {};
    v16bf a_cur = *(const v16bf*)ap;
#pragma unroll
    for (int k0 = 0; k0 < K; k0 += 32) {
        v16bf a_nxt = a_cur;
        if (k0 + 32 < K) a_nxt = *(const v16bf*)(ap + k0 + 32);
        v16bf b0 = *(const v16bf*)(sb0 + k0);
        v16bf b1 = *(const v16bf*)(sb1 + k0);
        v16bf b2 = *(const v16bf*)(sb2 + k0);
        v16bf b3 = *(const v16bf*)(sb3 + k0);
        c0 = __builtin_amdgcn_wmma_f32_16x16x32_bf16(false, a_cur, false, b0, (short)0, c0, false, false);
        c1 = __builtin_amdgcn_wmma_f32_16x16x32_bf16(false, a_cur, false, b1, (short)0, c1, false, false);
        c2 = __builtin_amdgcn_wmma_f32_16x16x32_bf16(false, a_cur, false, b2, (short)0, c2, false, false);
        c3 = __builtin_amdgcn_wmma_f32_16x16x32_bf16(false, a_cur, false, b3, (short)0, c3, false, false);
        a_cur = a_nxt;
    }
    // C layout: N = lane&15, M = laneH*8 + vgpr
    float* fr = Fo + (size_t)(row0 + laneH * 8) * Nn + col0 + laneM;
#pragma unroll
    for (int v = 0; v < 8; ++v) {
        fr[(size_t)v * Nn +  0] = c0[v];
        fr[(size_t)v * Nn + 16] = c1[v];
        fr[(size_t)v * Nn + 32] = c2[v];
        fr[(size_t)v * Nn + 48] = c3[v];
    }
}

// ---------------- sequence-axis LN stats per (batch, channel) --------------
__global__ __launch_bounds__(256) void stats_kernel(const float* __restrict__ Fo,
                                                    const float* __restrict__ lnb,
                                                    float* __restrict__ shift,
                                                    float* __restrict__ rstd) {
    int t = blockIdx.x * blockDim.x + threadIdx.x;
    if (t >= 16 * 768) return;
    int v = t % 768;
    int b = t / 768;
    const float* p = Fo + (size_t)b * 1024 * 768 + v;
    float s = 0.f, s2 = 0.f;
    for (int i = 0; i < 1024; ++i) {
        float f = p[(size_t)i * 768];
        s += f; s2 += f * f;
    }
    float mean = s  * (1.0f / 1024.0f);
    float msq  = s2 * (1.0f / 1024.0f);
    float var  = msq - mean * mean;
    float beta = lnb[v];
    shift[t] = beta - mean;
    rstd[t]  = rsqrtf(var + beta * beta + 0.001f);
}

// ---------------- normalize + tanh-GELU + cvt-to-bf16 ----------------------
__global__ __launch_bounds__(256) void normgelu_kernel(const float* __restrict__ Fo,
                                                       const float* __restrict__ shift,
                                                       const float* __restrict__ rstd,
                                                       unsigned short* __restrict__ G) {
    size_t t = (size_t)blockIdx.x * blockDim.x + threadIdx.x;
    const size_t TOT = (size_t)16384 * 768;
    if (t >= TOT) return;
    int v = (int)(t % 768);
    int b = (int)((t / 768) >> 10);
    int ch = b * 768 + v;
    float x = (Fo[t] + shift[ch]) * rstd[ch];
    float g = 0.5f * x * (1.0f + tanhf(0.7978845608028654f * (x + 0.044715f * x * x * x)));
    G[t] = f2bf(g);
}

// ---------------- GEMM2 + residual mix + permutation scatter ---------------
// mode 0: dest = rol(p) (seq = seq[:,ror]); mode 1: dest = ror(p);
// mode 2: identity; mode 3: final -> d_out at zflat(p)
__global__ __launch_bounds__(256) void gemm2_bf16(const unsigned short* __restrict__ G,
                                                  const unsigned short* __restrict__ Bt,
                                                  const float* __restrict__ seqCur,
                                                  const float* __restrict__ b2,
                                                  const float* __restrict__ rs,
                                                  float* __restrict__ outF,
                                                  unsigned short* __restrict__ outBf,
                                                  int mode) {
    extern __shared__ unsigned short sB[];               // [64][768]
    const int K = 768, Nn = 384, NT = Nn / 64;           // 6 n-groups
    int tid  = threadIdx.x;
    int lane = tid & 31, wv = tid >> 5;
    int mt8  = blockIdx.x / NT, ng = blockIdx.x % NT;
    int col0 = ng * 64;

    stage_weights(Bt + (size_t)col0 * K, sB, K, tid);

    int mt   = mt8 * 8 + wv;
    int row0 = mt * 16;
    int laneM = lane & 15, laneH = lane >> 4;

    const unsigned short* ap  = G  + (size_t)(row0 + laneM) * K + laneH * 16;
    const unsigned short* sb0 = sB + (size_t)( 0 + laneM) * K + laneH * 16;
    const unsigned short* sb1 = sB + (size_t)(16 + laneM) * K + laneH * 16;
    const unsigned short* sb2 = sB + (size_t)(32 + laneM) * K + laneH * 16;
    const unsigned short* sb3 = sB + (size_t)(48 + laneM) * K + laneH * 16;

    v8f c0 = {}, c1 = {}, c2 = {}, c3 = {};
    v16bf a_cur = *(const v16bf*)ap;
#pragma unroll
    for (int k0 = 0; k0 < K; k0 += 32) {
        v16bf a_nxt = a_cur;
        if (k0 + 32 < K) a_nxt = *(const v16bf*)(ap + k0 + 32);
        v16bf b0 = *(const v16bf*)(sb0 + k0);
        v16bf b1 = *(const v16bf*)(sb1 + k0);
        v16bf b2f = *(const v16bf*)(sb2 + k0);
        v16bf b3 = *(const v16bf*)(sb3 + k0);
        c0 = __builtin_amdgcn_wmma_f32_16x16x32_bf16(false, a_cur, false, b0,  (short)0, c0, false, false);
        c1 = __builtin_amdgcn_wmma_f32_16x16x32_bf16(false, a_cur, false, b1,  (short)0, c1, false, false);
        c2 = __builtin_amdgcn_wmma_f32_16x16x32_bf16(false, a_cur, false, b2f, (short)0, c2, false, false);
        c3 = __builtin_amdgcn_wmma_f32_16x16x32_bf16(false, a_cur, false, b3,  (short)0, c3, false, false);
        a_cur = a_nxt;
    }

    v8f cs[4] = {c0, c1, c2, c3};
#pragma unroll
    for (int j = 0; j < 4; ++j) {
        int n = col0 + j * 16 + laneM;       // output channel 0..383
        float bias = b2[n];
        float sg   = 1.0f / (1.0f + expf(-rs[n]));
        int   s4   = n / 96;                 // sub-position within 4-group
        int   u    = n - s4 * 96;            // feature index
#pragma unroll
        for (int v = 0; v < 8; ++v) {
            int r   = row0 + laneH * 8 + v;  // global row = b*1024 + blk
            int b   = r >> 10;
            int blk = r & 1023;
            float h   = seqCur[(size_t)r * 384 + n];
            float val = (cs[j][v] + bias) * CAND_W_F + sg * h;
            int p = (blk << 2) + s4;         // source sequence position
            if (mode == 3) {
                outF[((size_t)b * 4096 + zflat(p)) * 96 + u] = val;
            } else {
                int q = (mode == 0) ? (((p << 2) & 4095) | (p >> 10))
                      : (mode == 1) ? ((p >> 2) | ((p & 3) << 10))
                      : p;
                size_t di = (size_t)b * (1024 * 384) + (size_t)(q >> 2) * 384 + (q & 3) * 96 + u;
                outF[di]  = val;
                outBf[di] = f2bf(val);
            }
        }
    }
}

// ---------------------------------------------------------------------------
extern "C" void kernel_launch(void* const* d_in, const int* in_sizes, int n_in,
                              void* d_out, int out_size, void* d_ws, size_t ws_size,
                              hipStream_t stream) {
    const float* x   = (const float*)d_in[0];   // (16,64,64,96)
    const float* W1  = (const float*)d_in[1];   // (2,3,384,768)
    const float* lnb = (const float*)d_in[2];   // (2,3,768)
    const float* W2  = (const float*)d_in[3];   // (2,3,768,384)
    const float* b2  = (const float*)d_in[4];   // (2,3,384)
    const float* rsc = (const float*)d_in[5];   // (2,3,384)
    float* out = (float*)d_out;

    // ---- workspace carve-up (256B-aligned) ----
    char* ws = (char*)d_ws;
    size_t off = 0;
    auto carve = [&](size_t bytes) { void* p = ws + off; off += (bytes + 255) & ~(size_t)255; return p; };
    const size_t SEQ_F32 = (size_t)16384 * 384 * 4;
    const size_t SEQ_BF  = (size_t)16384 * 384 * 2;
    const size_t F_F32   = (size_t)16384 * 768 * 4;
    const size_t G_BF    = (size_t)16384 * 768 * 2;

    float*          seqF[2]; unsigned short* seqB[2];
    seqF[0] = (float*)carve(SEQ_F32);
    seqF[1] = (float*)carve(SEQ_F32);
    seqB[0] = (unsigned short*)carve(SEQ_BF);
    seqB[1] = (unsigned short*)carve(SEQ_BF);
    float*          Fbuf   = (float*)carve(F_F32);
    unsigned short* Gbuf   = (unsigned short*)carve(G_BF);
    float*          shiftb = (float*)carve((size_t)16 * 768 * 4);
    float*          rstdb  = (float*)carve((size_t)16 * 768 * 4);
    unsigned short* W1T    = (unsigned short*)carve((size_t)6 * 768 * 384 * 2);
    unsigned short* W2T    = (unsigned short*)carve((size_t)6 * 384 * 768 * 2);

    // ---- one-time prep ----
    prep_w1<<<6912, 256, 0, stream>>>(W1, W1T);
    prep_w2<<<6912, 256, 0, stream>>>(W2, W2T);
    init_perm<<<24576, 256, 0, stream>>>(x, seqF[0], seqB[0]);

    const size_t LDS1 = (size_t)64 * 384 * 2;   // 48 KB
    const size_t LDS2 = (size_t)64 * 768 * 2;   // 96 KB

    // ---- 22 switch layers ----
    int cur = 0;
    auto run_switch = [&](int bc, int s, int mode) {
        int g = bc * 3 + s;
        const unsigned short* w1t = W1T + (size_t)g * 768 * 384;
        const unsigned short* w2t = W2T + (size_t)g * 384 * 768;
        const float* lnb_s = lnb + (size_t)g * 768;
        const float* b2_s  = b2  + (size_t)g * 384;
        const float* rs_s  = rsc + (size_t)g * 384;

        gemm1_bf16<<<1536, 256, LDS1, stream>>>(seqB[cur], w1t, Fbuf);
        stats_kernel<<<48, 256, 0, stream>>>(Fbuf, lnb_s, shiftb, rstdb);
        normgelu_kernel<<<49152, 256, 0, stream>>>(Fbuf, shiftb, rstdb, Gbuf);

        int nxt = cur ^ 1;
        float* outF = (mode == 3) ? out : seqF[nxt];
        unsigned short* outBf = (mode == 3) ? nullptr : seqB[nxt];
        gemm2_bf16<<<768, 256, LDS2, stream>>>(Gbuf, w2t, seqF[cur], b2_s, rs_s,
                                               outF, outBf, mode);
        if (mode != 3) cur = nxt;
    };

    for (int bc = 0; bc < 2; ++bc) {
        for (int i = 0; i < 5; ++i) run_switch(bc, 0, 0);   // switch then seq[:,ror]
        for (int i = 0; i < 5; ++i) run_switch(bc, 1, 1);   // switch then seq[:,rol]
        run_switch(bc, 2, (bc == 1) ? 3 : 2);               // final switch of block
    }
    (void)in_sizes; (void)n_in; (void)out_size; (void)ws_size;
}
